// InterframeAligner_53678501265642
// MI455X (gfx1250) — matrix operand order repaired and checked by reading, not compiled, not run on hardware
//
#include <hip/hip_runtime.h>

// ---------------------------------------------------------------------------
// MI455X (gfx1250): padded-HWC implicit-GEMM convs + block attention, all
// matrix math on v_wmma_f32_16x16x32_bf16 (wave32, f32 accumulate).
//
// Conv activations live in zero-padded HWC buffers [b][258][258][C]; with
// K ordered tap-major (k = t*C + c, C % 16 == 0) every WMMA B fragment is a
// single contiguous 32B global load and the SAME-padding zeros come from the
// buffer border -> conv inner loop = pure global_load_b128 + v_wmma, no LDS,
// no bounds checks (same shape as the QK loop the compiler already pipelines
// with s_clause'd loads).
// ---------------------------------------------------------------------------

typedef __attribute__((ext_vector_type(16))) __bf16 v16bf;
typedef __attribute__((ext_vector_type(8)))  __bf16 v8bf;
typedef __attribute__((ext_vector_type(8)))  float  v8f;

#define IMG 256
#define PAD 258   // 256 + 1-pixel zero border on each side

__device__ __forceinline__ unsigned short f2bf(float f) {
    union { float f; unsigned u; } v; v.f = f;
    unsigned r = v.u + 0x7FFFu + ((v.u >> 16) & 1u);   // round-to-nearest-even
    return (unsigned short)(r >> 16);
}
__device__ __forceinline__ float bf2f(unsigned short h) {
    union { unsigned u; float f; } v; v.u = ((unsigned)h) << 16;
    return v.f;
}
// 16-element bf16 fragment from two 16B chunks (p0[0..7], p1[0..7]).
__device__ __forceinline__ v16bf load_frag(const unsigned short* p0,
                                           const unsigned short* p1) {
    v8bf lo = *reinterpret_cast<const v8bf*>(p0);
    v8bf hi = *reinterpret_cast<const v8bf*>(p1);
    v16bf r;
#pragma unroll
    for (int i = 0; i < 8; i++) { r[i] = lo[i]; r[8 + i] = hi[i]; }
    return r;
}

// ---------------------------------------------------------------------------
// Padded-HWC implicit-GEMM conv.
//   in : bf16 [8][PAD][PAD][CIN], zero border    (CIN % 16 == 0)
//   wp : bf16 [O][K],  K = CIN*TAPS, k = t*CIN + c
//   out: bf16 [8][PAD][PAD][O], interior written
// Block = (b, y, 128-px x segment); 8 waves, wave w owns pixels x0+16w..+15.
// Fragment k order: k = 16*(lane/16) + s (contiguous) in both A and B.
// ---------------------------------------------------------------------------
template <int MT, int CIN, int TAPS>
__global__ __launch_bounds__(256)
void conv_hwc(const unsigned short* __restrict__ in,
              const unsigned short* __restrict__ wp,
              const float*          __restrict__ bias,
              const unsigned short* __restrict__ res,   // padded [.. ][O] or null
              unsigned short*       __restrict__ out,
              int do_relu)
{
    constexpr int K = CIN * TAPS;      // multiple of 32
    constexpr int O = MT * 16;

    const int blk  = blockIdx.x;
    const int xseg = blk & 1;
    const int y    = (blk >> 1) & 255;
    const int b    = blk >> 9;
    const int tid  = threadIdx.x;
    const int lane = tid & 31;
    const int wave = tid >> 5;
    const int lrow = lane & 15;
    const int hi16 = (lane >= 16);
    const int x    = xseg * 128 + wave * 16 + lrow;   // this lane's pixel col

    const unsigned short* inb = in + (size_t)b * PAD * PAD * CIN;

    v8f acc[MT];
#pragma unroll
    for (int m = 0; m < MT; m++) {
#pragma unroll
        for (int r = 0; r < 8; r++) acc[m][r] = 0.f;
    }

    for (int kk = 0; kk < K; kk += 32) {
        // B fragment: 16 contiguous k = one tap, 16 consecutive channels.
        const int k0 = kk + (hi16 ? 16 : 0);
        const int t  = k0 / CIN;
        const int c0 = k0 - t * CIN;
        const int ty = (TAPS == 9) ? (t / 3) : 1;     // padded row offset
        const int tx = (TAPS == 9) ? (t % 3) : 1;     // padded col offset
        const unsigned short* bb =
            inb + ((size_t)(y + ty) * PAD + (x + tx)) * CIN + c0;
        v16bf bf = load_frag(bb, bb + 8);

#pragma unroll
        for (int m = 0; m < MT; m++) {
            const unsigned short* ab =
                wp + (size_t)(m * 16 + lrow) * K + kk + (hi16 ? 16 : 0);
            v16bf af = load_frag(ab, ab + 8);
            acc[m] = __builtin_amdgcn_wmma_f32_16x16x32_bf16(
                false, af, false, bf, (short)0, acc[m], false, false);
        }
    }

    // Epilogue: HWC => each lane's 8 accs are 8 consecutive channels -> one
    // 16B store (and one 16B residual load) per M tile.
    const size_t pixoff = (((size_t)b * PAD + (y + 1)) * PAD + (x + 1)) * O;
    unsigned short* op = out + pixoff;
#pragma unroll
    for (int m = 0; m < MT; m++) {
        const int ob = m * 16 + (hi16 ? 8 : 0);
        union { v8bf v; unsigned short s[8]; } sv, rv;
        if (res) rv.v = *reinterpret_cast<const v8bf*>(res + pixoff + ob);
#pragma unroll
        for (int r = 0; r < 8; r++) {
            float v = acc[m][r] + bias[ob + r];
            if (res) v += bf2f(rv.s[r]);
            if (do_relu) v = v > 0.f ? v : 0.f;
            sv.s[r] = f2bf(v);
        }
        *reinterpret_cast<v8bf*>(op + ob) = sv.v;
    }
}

// ---------------------------------------------------------------------------
// S[b][n][m] = scale * sum_d Q[b][n][d] * K[b][m][d]   (d = 2048)
// Block: 32 n x 128 m; both fragments are contiguous global gathers.
// ---------------------------------------------------------------------------
__global__ __launch_bounds__(256)
void qk_wmma(const unsigned short* __restrict__ Q,
             const unsigned short* __restrict__ Km,
             float* __restrict__ S, float scale)
{
    const int blk  = blockIdx.x;
    const int mblk = blk & 7;
    const int nblk = (blk >> 3) & 31;
    const int b    = blk >> 8;
    const int tid  = threadIdx.x, lane = tid & 31, wave = tid >> 5;
    const int lrow = lane & 15;
    const int hi16 = (lane >= 16);
    const int n0   = nblk * 32;
    const int m0   = mblk * 128 + wave * 16;
    const unsigned short* Qb = Q  + (size_t)b * 1024 * 2048;
    const unsigned short* Kb = Km + (size_t)b * 1024 * 2048;

    v8f acc[2];
#pragma unroll
    for (int m = 0; m < 2; m++) {
#pragma unroll
        for (int r = 0; r < 8; r++) acc[m][r] = 0.f;
    }

    for (int kk = 0; kk < 2048; kk += 32) {
        const unsigned short* bb =
            Kb + (size_t)(m0 + lrow) * 2048 + kk + (hi16 ? 16 : 0);
        v16bf bf = load_frag(bb, bb + 8);
#pragma unroll
        for (int m = 0; m < 2; m++) {
            const unsigned short* ab =
                Qb + (size_t)(n0 + m * 16 + lrow) * 2048 + kk + (hi16 ? 16 : 0);
            v16bf af = load_frag(ab, ab + 8);
            acc[m] = __builtin_amdgcn_wmma_f32_16x16x32_bf16(
                false, af, false, bf, (short)0, acc[m], false, false);
        }
    }

    float* Sb = S + (size_t)b * 1024 * 1024;
    const int mc = m0 + lrow;
    const int nb = hi16 ? 8 : 0;
#pragma unroll
    for (int m = 0; m < 2; m++) {
#pragma unroll
        for (int r = 0; r < 8; r++)
            Sb[(size_t)(n0 + m * 16 + nb + r) * 1024 + mc] = acc[m][r] * scale;
    }
}

// ---------------------------------------------------------------------------
// Row softmax over m (1024); writes attn f32 to d_out and bf16 copy for AV.
// ---------------------------------------------------------------------------
__global__ __launch_bounds__(256)
void softmax_row(const float* __restrict__ S,
                 float* __restrict__ attn_f,
                 unsigned short* __restrict__ attn_b)
{
    __shared__ float red[8];
    __shared__ float sval;
    const int row = blockIdx.x;           // b*1024 + n
    const int tid = threadIdx.x;
    const float* s = S + (size_t)row * 1024;
    float v[4];
    float mx = -3.4028235e38f;
#pragma unroll
    for (int i = 0; i < 4; i++) { v[i] = s[tid + 256 * i]; mx = fmaxf(mx, v[i]); }
#pragma unroll
    for (int o = 16; o; o >>= 1) mx = fmaxf(mx, __shfl_xor(mx, o, 32));
    if ((tid & 31) == 0) red[tid >> 5] = mx;
    __syncthreads();
    if (tid == 0) {
        float m = red[0];
        for (int i = 1; i < 8; i++) m = fmaxf(m, red[i]);
        sval = m;
    }
    __syncthreads();
    mx = sval;
    float sum = 0.f;
#pragma unroll
    for (int i = 0; i < 4; i++) { v[i] = __expf(v[i] - mx); sum += v[i]; }
#pragma unroll
    for (int o = 16; o; o >>= 1) sum += __shfl_xor(sum, o, 32);
    __syncthreads();
    if ((tid & 31) == 0) red[tid >> 5] = sum;
    __syncthreads();
    if (tid == 0) {
        float m = 0.f;
        for (int i = 0; i < 8; i++) m += red[i];
        sval = m;
    }
    __syncthreads();
    const float rs = 1.f / sval;
    float* of = attn_f + (size_t)row * 1024;
    unsigned short* ob = attn_b + (size_t)row * 1024;
#pragma unroll
    for (int i = 0; i < 4; i++) {
        float o = v[i] * rs;
        of[tid + 256 * i] = o;
        ob[tid + 256 * i] = f2bf(o);
    }
}

// ---------------------------------------------------------------------------
// recons[b][n][e] = sum_m attn[b][n][m] * V[b][m][e]; fused tanh +
// blocks_to_tensor scatter. Vt stored [b][e][m] -> contiguous B fragments.
// ---------------------------------------------------------------------------
__global__ __launch_bounds__(256)
void av_wmma(const unsigned short* __restrict__ A,   // attn bf16 [b][1024][1024]
             const unsigned short* __restrict__ Vt,  // bf16 [b][192][1024]
             float* __restrict__ outT)               // f32 [8][3][256][256]
{
    const int blk  = blockIdx.x;
    const int b    = blk / 96;
    const int t    = blk - b * 96;
    const int tid  = threadIdx.x, lane = tid & 31, wave = tid >> 5;
    const int wt   = t * 8 + wave;          // 0..767 = 64 ntiles x 12 etiles
    const int ntile = wt / 12;
    const int etile = wt - ntile * 12;
    const int lrow = lane & 15;
    const int hi16 = (lane >= 16);
    const unsigned short* Ab = A  + (size_t)b * 1024 * 1024;
    const unsigned short* Vb = Vt + (size_t)b * 192 * 1024;

    v8f acc;
#pragma unroll
    for (int r = 0; r < 8; r++) acc[r] = 0.f;

    for (int kk = 0; kk < 1024; kk += 32) {
        const unsigned short* bb =
            Vb + (size_t)(etile * 16 + lrow) * 1024 + kk + (hi16 ? 16 : 0);
        v16bf bf = load_frag(bb, bb + 8);
        const unsigned short* ab =
            Ab + (size_t)(ntile * 16 + lrow) * 1024 + kk + (hi16 ? 16 : 0);
        v16bf af = load_frag(ab, ab + 8);
        acc = __builtin_amdgcn_wmma_f32_16x16x32_bf16(
            false, af, false, bf, (short)0, acc, false, false);
    }

    const int e  = etile * 16 + lrow;
    const int c  = e >> 6, iy = (e >> 3) & 7, ix = e & 7;
    const int nb = hi16 ? 8 : 0;
#pragma unroll
    for (int r = 0; r < 8; r++) {
        const int n  = ntile * 16 + nb + r;
        const int by = n >> 5, bx = n & 31;
        outT[(((size_t)b * 3 + c) * IMG + by * 8 + iy) * IMG + bx * 8 + ix] =
            tanhf(acc[r]);
    }
}

// ---------------------------------------------------------------------------
// Data-movement kernels.
// ---------------------------------------------------------------------------
// frame f32 NCHW -> padded HWC bf16 [b][PAD][PAD][32], channels 3..31 = 0.
__global__ __launch_bounds__(256)
void cvt_pad_k(const float* __restrict__ fr, unsigned short* __restrict__ dst) {
    int i = blockIdx.x * 256 + threadIdx.x;   // 8*256*256*32 = 16,777,216
    int c = i & 31;
    int x = (i >> 5) & 255;
    int y = (i >> 13) & 255;
    int b = i >> 21;
    unsigned short v = 0;
    if (c < 3) v = f2bf(fr[(((size_t)b * 3 + c) * IMG + y) * IMG + x]);
    dst[(((size_t)b * PAD + (y + 1)) * PAD + (x + 1)) * 32 + c] = v;
}

// w [O][Creal][taps] (OIHW flat) -> wp[o][t*Cpad + c] bf16, zero-padded chans.
__global__ __launch_bounds__(256)
void pack_w_k(const float* __restrict__ w, unsigned short* __restrict__ wp,
              int O, int Creal, int Cpad, int taps) {
    int K = Cpad * taps;
    int i = blockIdx.x * 256 + threadIdx.x;
    if (i >= O * K) return;
    int o = i / K, k = i - o * K;
    int t = k / Cpad, c = k - t * Cpad;
    unsigned short v = 0;
    if (c < Creal) v = f2bf(w[(o * Creal + c) * taps + t]);
    wp[i] = v;
}

// featP padded HWC [b][PAD][PAD][32] -> blocked [b][n][d], d = c*64+iy*8+ix.
__global__ __launch_bounds__(256)
void repack_qk_k(const unsigned short* __restrict__ featP,
                 unsigned short* __restrict__ M) {
    int i = blockIdx.x * 256 + threadIdx.x;   // 8*1024*2048 = 16,777,216
    int d = i & 2047;
    int nbi = i >> 11;
    int n = nbi & 1023;
    int b = nbi >> 10;
    int c = d >> 6, iy = (d >> 3) & 7, ix = d & 7;
    int by = n >> 5, bx = n & 31;
    M[i] = featP[(((size_t)b * PAD + (by * 8 + iy + 1)) * PAD +
                  (bx * 8 + ix + 1)) * 32 + c];
}

// frame_tm1 f32 -> Vt bf16 [b][e][m] (transposed so AV B-frags contiguous).
__global__ __launch_bounds__(256)
void build_vt_k(const float* __restrict__ fr, unsigned short* __restrict__ Vt) {
    int i = blockIdx.x * 256 + threadIdx.x;   // 8*192*1024 = 1,572,864
    int m = i & 1023;
    int be = i >> 10;
    int e = be % 192, b = be / 192;
    int c = e >> 6, iy = (e >> 3) & 7, ix = e & 7;
    int by = m >> 5, bx = m & 31;
    Vt[i] = f2bf(fr[(((size_t)b * 3 + c) * IMG + by * 8 + iy) * IMG + bx * 8 + ix]);
}

// ---------------------------------------------------------------------------
extern "C" void kernel_launch(void* const* d_in, const int* in_sizes, int n_in,
                              void* d_out, int out_size, void* d_ws, size_t ws_size,
                              hipStream_t stream)
{
    const float* frame_tm1 = (const float*)d_in[0];
    const float* frame_t   = (const float*)d_in[1];
    const float* c0w  = (const float*)d_in[2];  const float* c0b  = (const float*)d_in[3];
    const float* r1w1 = (const float*)d_in[4];  const float* r1b1 = (const float*)d_in[5];
    const float* r1w2 = (const float*)d_in[6];  const float* r1b2 = (const float*)d_in[7];
    const float* r1sw = (const float*)d_in[8];  const float* r1sb = (const float*)d_in[9];
    const float* r2w1 = (const float*)d_in[10]; const float* r2b1 = (const float*)d_in[11];
    const float* r2w2 = (const float*)d_in[12]; const float* r2b2 = (const float*)d_in[13];
    const float* cow  = (const float*)d_in[14]; const float* cob  = (const float*)d_in[15];

    // ---- workspace carve, 256B-aligned slices ----
    const size_t P32 = (size_t)8 * PAD * PAD * 32 * 2;   // 34,080,768 B
    const size_t P64 = (size_t)8 * PAD * PAD * 64 * 2;   // 68,161,536 B
    char* ws = (char*)d_ws;
    size_t off = 0;
    auto carve = [&](size_t bytes) -> char* {
        char* p = ws + off; off += (bytes + 255) & ~(size_t)255; return p;
    };
    unsigned short* IN32P   = (unsigned short*)carve(P32);  // padded input, reused per frame
    unsigned short* ACT32P  = (unsigned short*)carve(P32);  // h32
    unsigned short* A64PA   = (unsigned short*)carve(P64);
    unsigned short* A64PB   = (unsigned short*)carve(P64);
    unsigned short* A64PC   = (unsigned short*)carve(P64);
    unsigned short* FEATP_T = (unsigned short*)carve(P32);
    unsigned short* FEATP_M = (unsigned short*)carve(P32);
    unsigned short* VT      = (unsigned short*)carve(1572864ull * 2);
    unsigned short* WPC0    = (unsigned short*)carve(32 * 288 * 2);
    unsigned short* WPR1W1  = (unsigned short*)carve(64 * 288 * 2);
    unsigned short* WPR1SW  = (unsigned short*)carve(64 * 32  * 2);
    unsigned short* WPR1W2  = (unsigned short*)carve(64 * 576 * 2);
    unsigned short* WPR2W1  = (unsigned short*)carve(64 * 576 * 2);
    unsigned short* WPR2W2  = (unsigned short*)carve(64 * 576 * 2);
    unsigned short* WPCOW   = (unsigned short*)carve(32 * 64  * 2);
    // reuse freed conv scratch after the conv stage:
    unsigned short* Qm    = A64PA;          // bf16 [8][1024][2048] (33.5 MB)
    unsigned short* Kmat  = A64PB;          // bf16 [8][1024][2048]
    float*          Sf    = (float*)A64PC;  // f32  [8][1024][1024] (33.5 MB)
    unsigned short* ATTNB = ACT32P;         // bf16 [8][1024][1024] (16.8 MB)

    float* out_f  = (float*)d_out;          // reassembled [8][3][256][256]
    float* attn_f = out_f + 1572864;        // attn [8][1024][1024]

    // ---- stage 0: zero padded buffers (borders), pack weights, build V ----
    hipMemsetAsync(IN32P,   0, P32, stream);
    hipMemsetAsync(ACT32P,  0, P32, stream);
    hipMemsetAsync(A64PA,   0, P64, stream);
    hipMemsetAsync(A64PB,   0, P64, stream);
    hipMemsetAsync(A64PC,   0, P64, stream);
    hipMemsetAsync(FEATP_T, 0, P32, stream);
    hipMemsetAsync(FEATP_M, 0, P32, stream);
    pack_w_k<<<(32*288 + 255)/256, 256, 0, stream>>>(c0w,  WPC0,   32,  3, 32, 9);
    pack_w_k<<<(64*288 + 255)/256, 256, 0, stream>>>(r1w1, WPR1W1, 64, 32, 32, 9);
    pack_w_k<<<(64*32  + 255)/256, 256, 0, stream>>>(r1sw, WPR1SW, 64, 32, 32, 1);
    pack_w_k<<<(64*576 + 255)/256, 256, 0, stream>>>(r1w2, WPR1W2, 64, 64, 64, 9);
    pack_w_k<<<(64*576 + 255)/256, 256, 0, stream>>>(r2w1, WPR2W1, 64, 64, 64, 9);
    pack_w_k<<<(64*576 + 255)/256, 256, 0, stream>>>(r2w2, WPR2W2, 64, 64, 64, 9);
    pack_w_k<<<(32*64  + 255)/256, 256, 0, stream>>>(cow,  WPCOW,  32, 64, 64, 1);
    build_vt_k<<<6144, 256, 0, stream>>>(frame_tm1, VT);

    // ---- stage 1: feat_net on both frames (pure load+WMMA inner loops) ----
    for (int f = 0; f < 2; f++) {
        const float*    frame = (f == 0) ? frame_t : frame_tm1;
        unsigned short* featP = (f == 0) ? FEATP_T : FEATP_M;
        cvt_pad_k<<<65536, 256, 0, stream>>>(frame, IN32P);
        conv_hwc<2,32,9><<<4096,256,0,stream>>>(IN32P,  WPC0,   c0b,  nullptr, ACT32P, 1); // h32
        conv_hwc<4,32,9><<<4096,256,0,stream>>>(ACT32P, WPR1W1, r1b1, nullptr, A64PA,  1); // y1
        conv_hwc<4,32,1><<<4096,256,0,stream>>>(ACT32P, WPR1SW, r1sb, nullptr, A64PB,  0); // skip
        conv_hwc<4,64,9><<<4096,256,0,stream>>>(A64PA,  WPR1W2, r1b2, A64PB,   A64PC,  1); // h64
        conv_hwc<4,64,9><<<4096,256,0,stream>>>(A64PC,  WPR2W1, r2b1, nullptr, A64PA,  1); // y2
        conv_hwc<4,64,9><<<4096,256,0,stream>>>(A64PA,  WPR2W2, r2b2, A64PC,   A64PB,  1); // h64b
        conv_hwc<2,64,1><<<4096,256,0,stream>>>(A64PB,  WPCOW,  cob,  nullptr, featP,  0); // feat
    }

    // ---- stage 2: block attention ----
    repack_qk_k<<<65536, 256, 0, stream>>>(FEATP_T, Qm);
    repack_qk_k<<<65536, 256, 0, stream>>>(FEATP_M, Kmat);
    const float scale = 0.022097086912079608f;   // 1/sqrt(2048)
    qk_wmma<<<2048, 256, 0, stream>>>(Qm, Kmat, Sf, scale);
    softmax_row<<<8192, 256, 0, stream>>>(Sf, attn_f, ATTNB);
    av_wmma<<<768, 256, 0, stream>>>(ATTNB, VT, out_f);

    (void)in_sizes; (void)n_in; (void)out_size; (void)ws_size;
}
